// DynamicMaskAttention_20864951124271
// MI455X (gfx1250) — matrix operand order repaired and checked by reading, not compiled, not run on hardware
//
#include <hip/hip_runtime.h>
#include <hip/hip_bf16.h>
#include <stdint.h>

#define B_   2
#define S_   2048
#define D_   2048
#define H_   16
#define HKV_ 4
#define HD_  128
#define G_   4

typedef __attribute__((ext_vector_type(16))) __bf16          v16bf;
typedef __attribute__((ext_vector_type(16))) unsigned short  v16us;
typedef __attribute__((ext_vector_type(8)))  unsigned short  v8us;
typedef __attribute__((ext_vector_type(8)))  float           v8f;
typedef __attribute__((ext_vector_type(4)))  unsigned int    v4u;
typedef __attribute__((ext_vector_type(8)))  unsigned int    v8u;

static __device__ __forceinline__ unsigned short f2bf(float f) {
  unsigned int u = __float_as_uint(f);
  u += 0x7FFFu + ((u >> 16) & 1u);            // round-to-nearest-even
  return (unsigned short)(u >> 16);
}
static __device__ __forceinline__ float bf2f(unsigned short h) {
  return __uint_as_float(((unsigned int)h) << 16);
}

// Assemble a 16-element bf16 fragment from two contiguous 8-element (16B) loads.
static __device__ __forceinline__ v16us combine8(const unsigned short* p0,
                                                 const unsigned short* p1) {
  v8us a = *(const v8us*)p0;
  v8us b = *(const v8us*)p1;
  v16us r;
#pragma unroll
  for (int i = 0; i < 8; ++i) { r[i] = a[i]; r[i + 8] = b[i]; }
  return r;
}

static __device__ __forceinline__ v8f wmma_bf16(v16us a, v16us b, v8f c) {
  return __builtin_amdgcn_wmma_f32_16x16x32_bf16(
      false, __builtin_bit_cast(v16bf, a),
      false, __builtin_bit_cast(v16bf, b),
      (short)0, c, false, false);
}

// A-fragment (16x32 bf16), saddr-friendly: uniform base + invariant u32 index.
// Index already contains row*K + 8*half; elements at idx..idx+7 and idx+16..+23.
static __device__ __forceinline__ v16us loadAs(const unsigned short* base, unsigned idx) {
  const unsigned short* p = base + idx;
  return combine8(p, p + 16);
}
// B-fragment (32x16 bf16) from BT[N,K]: index contains n_row*K + 8*half.
static __device__ __forceinline__ v16us loadBs(const unsigned short* base, unsigned idx) {
  const unsigned short* p = base + idx;
  return combine8(p, p + 16);
}
// Pointer-based variants (used where the base itself is per-lane, e.g. LDS rows).
static __device__ __forceinline__ v16us loadA(const unsigned short* row, int k, int half) {
  const unsigned short* p = row + k + 8 * half;
  return combine8(p, p + 16);
}
static __device__ __forceinline__ v16us loadB(const unsigned short* row, int k, int half) {
  const unsigned short* p = row + k + 8 * half;
  return combine8(p, p + 16);
}

// Raw LDS byte address of a __shared__ object (generic -> AS(3) strips aperture).
static __device__ __forceinline__ unsigned lds_addr_of(const void* p) {
  return (unsigned)(size_t)(__attribute__((address_space(3))) const void*)p;
}

// Issue a 2-D TENSOR_LOAD_TO_LDS via the Tensor Data Mover.
// D# packing per CDNA5 ISA (2 SGPR groups, 2-D tile).
// All args wave-uniform. Completion tracked with TENSORcnt.
static __device__ __forceinline__ void tdm_load_2d(
    unsigned lds_byte_addr, const void* gaddr,
    unsigned tensor_d0,   // row length in elements (OOB bound)
    unsigned tensor_d1,   // row count (OOB bound)
    unsigned d0_stride,   // row stride in elements
    unsigned tile_d0,     // tile width in elements
    unsigned tile_d1)     // tile height in rows
{
  const unsigned long long ga = (unsigned long long)(size_t)gaddr;
  v4u g0;
  g0[0] = 1u;                                              // count=1, user D#
  g0[1] = lds_byte_addr;                                   // lds_addr
  g0[2] = (unsigned)(ga & 0xFFFFFFFFu);                    // global_addr[31:0]
  g0[3] = (unsigned)((ga >> 32) & 0x01FFFFFFu) | (2u << 30); // addr[56:32] | type=2
  v8u g1;
  g1[0] = (1u << 16);                                      // data_size=1 -> 2 bytes
  g1[1] = (tensor_d0 & 0xFFFFu) << 16;                     // tensor_dim0[15:0]
  g1[2] = (tensor_d0 >> 16) | ((tensor_d1 & 0xFFFFu) << 16); // dim0 hi | dim1 lo
  g1[3] = (tensor_d1 >> 16) | (tile_d0 << 16);             // dim1 hi | tile_dim0
  g1[4] = tile_d1 & 0xFFFFu;                               // tile_dim1 (tile_dim2=0)
  g1[5] = d0_stride;                                       // tensor_dim0_stride[31:0]
  g1[6] = 0u;                                              // stride hi | dim1_stride lo
  g1[7] = 0u;                                              // dim1_stride hi
  asm volatile("tensor_load_to_lds %0, %1" :: "s"(g0), "s"(g1) : "memory");
}

// ---------------------------------------------------------------------------
// fp32 -> bf16 bulk convert
__global__ void f32_to_bf16_kernel(const float* __restrict__ in,
                                   unsigned short* __restrict__ out, size_t n) {
  size_t i = (size_t)blockIdx.x * blockDim.x + threadIdx.x;
  if (i < n) out[i] = f2bf(in[i]);
}

// W[R,C] fp32 -> WT[C,R] bf16, LDS-tiled so both sides are coalesced.
__global__ __launch_bounds__(256) void transpose_f32_to_bf16_tiled(
    const float* __restrict__ in, unsigned short* __restrict__ out, int R, int C) {
  __shared__ unsigned short tile[32][33];
  const int tx = threadIdx.x, ty = threadIdx.y;
  const int r0 = blockIdx.y * 32;
  const int c0 = blockIdx.x * 32;
#pragma unroll
  for (int i = 0; i < 4; ++i)
    tile[ty + 8 * i][tx] = f2bf(in[(size_t)(r0 + ty + 8 * i) * C + c0 + tx]);
  __syncthreads();
#pragma unroll
  for (int i = 0; i < 4; ++i)
    out[(size_t)(c0 + ty + 8 * i) * R + r0 + tx] = tile[tx][ty + 8 * i];
}

// v[b,s,n,d] bf16 -> vT[b,n,d,s] bf16, LDS-tiled per (b,n) plane.
__global__ __launch_bounds__(256) void transpose_v_tiled(
    const unsigned short* __restrict__ vb, unsigned short* __restrict__ vT) {
  __shared__ unsigned short tile[32][33];
  const int tx = threadIdx.x, ty = threadIdx.y;
  const int bn = blockIdx.z;                 // b*HKV + n
  const int b = bn / HKV_, n = bn % HKV_;
  const int s0 = blockIdx.y * 32;
  const int d0 = blockIdx.x * 32;
  const unsigned short* src = vb + (size_t)b * S_ * (HKV_ * HD_) + (size_t)n * HD_;
#pragma unroll
  for (int i = 0; i < 4; ++i)
    tile[ty + 8 * i][tx] = src[(size_t)(s0 + ty + 8 * i) * (HKV_ * HD_) + d0 + tx];
  __syncthreads();
  unsigned short* dst = vT + (size_t)bn * HD_ * S_;
#pragma unroll
  for (int i = 0; i < 4; ++i)
    dst[(size_t)(d0 + ty + 8 * i) * S_ + s0 + tx] = tile[tx][ty + 8 * i];
}

// ---------------------------------------------------------------------------
// WMMA GEMM: C[M,N] = A[M,K] @ B[K,N]; A bf16 row-major, BT[N,K] bf16.
// 8 waves / 128x128 block tile; 32x64 per wave; register ping-pong over 64-K.
// Addressing: uniform base pointers (Ak/Bk, bumped by scalar adds) + per-lane
// loop-invariant u32 offsets -> saddr global_load_b128, no per-iteration VALU
// address math (avoids WMMA->VALU hazard nops). Final 64-K chunk peeled.
__global__ __launch_bounds__(256) void gemm_bf16_wmma(
    const unsigned short* __restrict__ A, const unsigned short* __restrict__ BT,
    void* __restrict__ C, int M, int N, int K, int c_f32) {
  const int wave = threadIdx.x >> 5;
  const int lane = threadIdx.x & 31;
  const int half = lane >> 4;
  const int lcol = lane & 15;
  const int wm = wave >> 1;                  // 0..3
  const int wn = wave & 1;                   // 0..1
  const int m0 = blockIdx.x * 128 + wm * 32;
  const int n0 = blockIdx.y * 128 + wn * 64;

  const v8f vzero = {0.f, 0.f, 0.f, 0.f, 0.f, 0.f, 0.f, 0.f};
  v8f acc[2][4];
#pragma unroll
  for (int i = 0; i < 2; ++i)
#pragma unroll
    for (int t = 0; t < 4; ++t) acc[i][t] = vzero;

  // Loop-invariant per-lane element offsets (all buffers < 2^31 bytes).
  unsigned aoffh[2], boffh[4];
  aoffh[0] = (unsigned)(m0 + lcol) * (unsigned)K + 8u * (unsigned)half;
  aoffh[1] = aoffh[0] + 16u * (unsigned)K;
#pragma unroll
  for (int t = 0; t < 4; ++t)
    boffh[t] = (unsigned)(n0 + 16 * t + lcol) * (unsigned)K + 8u * (unsigned)half;

  v16us a0[2], b0[4], a1[2], b1[4];
#pragma unroll
  for (int i = 0; i < 2; ++i) a0[i] = loadAs(A, aoffh[i]);
#pragma unroll
  for (int t = 0; t < 4; ++t) b0[t] = loadBs(BT, boffh[t]);

  const unsigned short* Ak = A;
  const unsigned short* Bk = BT;
  for (int k0 = 0; k0 + 64 < K; k0 += 64) {
    // stage 1: load k0+32 fragments, then WMMA on k0 fragments
#pragma unroll
    for (int i = 0; i < 2; ++i) a1[i] = loadAs(Ak + 32, aoffh[i]);
#pragma unroll
    for (int t = 0; t < 4; ++t) b1[t] = loadBs(Bk + 32, boffh[t]);
#pragma unroll
    for (int i = 0; i < 2; ++i)
#pragma unroll
      for (int t = 0; t < 4; ++t) acc[i][t] = wmma_bf16(a0[i], b0[t], acc[i][t]);

    // stage 2: load k0+64 fragments, then WMMA on k0+32 fragments
#pragma unroll
    for (int i = 0; i < 2; ++i) a0[i] = loadAs(Ak + 64, aoffh[i]);
#pragma unroll
    for (int t = 0; t < 4; ++t) b0[t] = loadBs(Bk + 64, boffh[t]);
#pragma unroll
    for (int i = 0; i < 2; ++i)
#pragma unroll
      for (int t = 0; t < 4; ++t) acc[i][t] = wmma_bf16(a1[i], b1[t], acc[i][t]);

    Ak += 64;
    Bk += 64;
  }
  // Peeled final 64-K chunk: only the +32 half still needs loading.
#pragma unroll
  for (int i = 0; i < 2; ++i) a1[i] = loadAs(Ak + 32, aoffh[i]);
#pragma unroll
  for (int t = 0; t < 4; ++t) b1[t] = loadBs(Bk + 32, boffh[t]);
#pragma unroll
  for (int i = 0; i < 2; ++i)
#pragma unroll
    for (int t = 0; t < 4; ++t) acc[i][t] = wmma_bf16(a0[i], b0[t], acc[i][t]);
#pragma unroll
  for (int i = 0; i < 2; ++i)
#pragma unroll
    for (int t = 0; t < 4; ++t) acc[i][t] = wmma_bf16(a1[i], b1[t], acc[i][t]);

#pragma unroll
  for (int i = 0; i < 2; ++i)
#pragma unroll
    for (int t = 0; t < 4; ++t)
#pragma unroll
      for (int r = 0; r < 8; ++r) {
        const unsigned row = (unsigned)(m0 + 16 * i + r + 8 * half);
        const unsigned col = (unsigned)(n0 + 16 * t + lcol);
        const float v = acc[i][t][r];
        if (c_f32) ((float*)C)[row * (unsigned)N + col] = v;
        else       ((unsigned short*)C)[row * (unsigned)N + col] = f2bf(v);
      }
}

// ---------------------------------------------------------------------------
// bias[b,n,s] = sigmoid(q_flat[b,s,:] @ Wg[:,n]) * (v_flat[b,s,:] @ Wd[:,n])
__global__ __launch_bounds__(256) void bias_kernel(
    const unsigned short* __restrict__ qb, const unsigned short* __restrict__ vb,
    const float* __restrict__ Wg, const float* __restrict__ Wd,
    float* __restrict__ bias) {
  const int wave = threadIdx.x >> 5;
  const int lane = threadIdx.x & 31;
  const int row = blockIdx.x * 8 + wave;      // [0, B_*S_)
  const int b = row / S_, s = row % S_;
  float g[HKV_] = {0.f, 0.f, 0.f, 0.f};
  float dl[HKV_] = {0.f, 0.f, 0.f, 0.f};
  const unsigned short* q = qb + (unsigned)row * (unsigned)(H_ * HD_);
  for (int i = lane; i < H_ * HD_; i += 32) {
    const float qv = bf2f(q[i]);
#pragma unroll
    for (int n = 0; n < HKV_; ++n) g[n] += qv * Wg[(unsigned)i * HKV_ + n];
  }
  const unsigned short* v = vb + (unsigned)row * (unsigned)(HKV_ * HD_);
  for (int i = lane; i < HKV_ * HD_; i += 32) {
    const float vv = bf2f(v[i]);
#pragma unroll
    for (int n = 0; n < HKV_; ++n) dl[n] += vv * Wd[(unsigned)i * HKV_ + n];
  }
#pragma unroll
  for (int n = 0; n < HKV_; ++n) {
#pragma unroll
    for (int off = 16; off > 0; off >>= 1) {
      g[n] += __shfl_xor(g[n], off, 32);
      dl[n] += __shfl_xor(dl[n], off, 32);
    }
  }
  if (lane == 0) {
#pragma unroll
    for (int n = 0; n < HKV_; ++n) {
      const float sg = 1.f / (1.f + __expf(-g[n]));
      bias[((size_t)b * HKV_ + n) * S_ + s] = sg * dl[n];
    }
  }
}

// ---------------------------------------------------------------------------
// Flash attention: one 4-wave block per (b, kv-head, 16-query tile); wave w
// handles query head h = n*G + w, so all G heads share the K/V LDS panels.
// K (32x128) and V^T (128x32) panels are double-buffered in LDS via the
// Tensor Data Mover; wave 0 issues TDM for block j+1, waits TENSORcnt<=2
// (TDM is in-order per wave), then a workgroup barrier publishes the panel.
__global__ __launch_bounds__(128) void fa_wmma_kernel(
    const unsigned short* __restrict__ qb,   // [B,S,H,HD] bf16
    const unsigned short* __restrict__ kb,   // [B,S,HKV,HD] bf16
    const unsigned short* __restrict__ vT,   // [B,HKV,HD,S] bf16
    const float* __restrict__ bias,          // [B,HKV,S]
    unsigned short* __restrict__ ob)         // [B,S,H,HD] bf16
{
  const int qt = blockIdx.x;
  const int n  = blockIdx.y;                 // kv head
  const int b  = blockIdx.z;
  const int wave = threadIdx.x >> 5;         // group member g
  const int lane = threadIdx.x & 31;
  const int h = n * G_ + wave;               // query head
  const int half = lane >> 4;
  const int lcol = lane & 15;
  const int q0 = qt * 16;
  const float scale = 0.08838834764831845f;  // HD^-0.5

  __shared__ __align__(16) unsigned short k_lds[2][32 * HD_];  // 2 x 8 KB
  __shared__ __align__(16) unsigned short v_lds[2][HD_ * 32];  // 2 x 8 KB
  __shared__ __align__(16) unsigned short p_lds[4][16 * 32];   // per-wave P tile

  const unsigned short* kbase = kb + ((size_t)b * S_ * HKV_ + n) * HD_;
  const unsigned short* vbase = vT + ((size_t)b * HKV_ + n) * HD_ * S_;

  // Q fragments (16 q-rows x 128 d) resident in registers.
  v16us qf[4];
  const unsigned qoff = ((unsigned)(b * S_ + q0 + lcol) * H_ + h) * HD_;
#pragma unroll
  for (int c = 0; c < 4; ++c) qf[c] = loadA(qb + qoff, 32 * c, half);

  const v8f vzero = {0.f, 0.f, 0.f, 0.f, 0.f, 0.f, 0.f, 0.f};
  v8f oa[8];
#pragma unroll
  for (int t = 0; t < 8; ++t) oa[t] = vzero;
  float mrow[8], lrow[8];
#pragma unroll
  for (int r = 0; r < 8; ++r) { mrow[r] = -3.0e38f; lrow[r] = 0.f; }

  const float* brow = bias + ((size_t)b * HKV_ + n) * S_;
  const int nblocks = (q0 + 16 + 31) / 32;

  if (wave == 0) {
    tdm_load_2d(lds_addr_of(&k_lds[0][0]), kbase, HD_, 32, HKV_ * HD_, HD_, 32);
    tdm_load_2d(lds_addr_of(&v_lds[0][0]), vbase, 32, HD_, S_, 32, HD_);
  }

  for (int j = 0; j < nblocks; ++j) {
    const int k0 = j * 32;
    const int cur = j & 1;
    if (wave == 0) {
      if (j + 1 < nblocks) {
        tdm_load_2d(lds_addr_of(&k_lds[1 - cur][0]),
                    kbase + (size_t)(k0 + 32) * (HKV_ * HD_),
                    HD_, 32, HKV_ * HD_, HD_, 32);
        tdm_load_2d(lds_addr_of(&v_lds[1 - cur][0]),
                    vbase + (k0 + 32), 32, HD_, S_, 32, HD_);
        __builtin_amdgcn_s_wait_tensorcnt(2);   // current block's 2 TDM ops done
      } else {
        __builtin_amdgcn_s_wait_tensorcnt(0);
      }
    }
    __syncthreads();                            // publish panel to all 4 waves

    // QK^T: 8 K-fragments from LDS, then 8 WMMAs.
    v16us kf[8];
#pragma unroll
    for (int c = 0; c < 4; ++c) {
      kf[c]     = loadB(&k_lds[cur][(unsigned)lcol * HD_],        32 * c, half);
      kf[4 + c] = loadB(&k_lds[cur][(unsigned)(16 + lcol) * HD_], 32 * c, half);
    }
    v8f sc0 = vzero, sc1 = vzero;
#pragma unroll
    for (int c = 0; c < 4; ++c) {
      sc0 = wmma_bf16(qf[c], kf[c], sc0);
      sc1 = wmma_bf16(qf[c], kf[4 + c], sc1);
    }

    const float bias0 = brow[k0 + lcol];
    const float bias1 = brow[k0 + 16 + lcol];
    const int key0 = k0 + lcol;
    const int key1 = k0 + 16 + lcol;

    float p0[8], p1[8];
#pragma unroll
    for (int r = 0; r < 8; ++r) {
      const int qpos = q0 + r + 8 * half;      // this lane's query row
      float v0 = (key0 <= qpos && bias0 > 0.f) ? sc0[r] * scale + bias0 : -3.0e38f;
      float v1 = (key1 <= qpos && bias1 > 0.f) ? sc1[r] * scale + bias1 : -3.0e38f;
      float mx = fmaxf(v0, v1);
#pragma unroll
      for (int off = 1; off < 16; off <<= 1)   // row max across 16 columns
        mx = fmaxf(mx, __shfl_xor(mx, off, 16));
      mx = fmaxf(mx, -1.0e30f);
      const float mnew = fmaxf(mrow[r], mx);
      const float alpha = __expf(mrow[r] - mnew);
      mrow[r] = mnew;
      p0[r] = __expf(v0 - mnew);
      p1[r] = __expf(v1 - mnew);
      float rs = p0[r] + p1[r];
#pragma unroll
      for (int off = 1; off < 16; off <<= 1) rs += __shfl_xor(rs, off, 16);
      lrow[r] = lrow[r] * alpha + rs;
#pragma unroll
      for (int t = 0; t < 8; ++t) oa[t][r] = oa[t][r] * alpha;
    }

    // Redistribute P from C-layout to A-layout through this wave's LDS tile.
#pragma unroll
    for (int r = 0; r < 8; ++r) {
      p_lds[wave][(r + 8 * half) * 32 + lcol]      = f2bf(p0[r]);
      p_lds[wave][(r + 8 * half) * 32 + 16 + lcol] = f2bf(p1[r]);
    }
    v16us pf = combine8(&p_lds[wave][lcol * 32 + 8 * half],
                        &p_lds[wave][lcol * 32 + 8 * half + 16]);

    // PV: 8 V-fragments from LDS, then 8 WMMAs.
    v16us vf[8];
#pragma unroll
    for (int t = 0; t < 8; ++t)
      vf[t] = loadB(&v_lds[cur][(unsigned)(16 * t + lcol) * 32], 0, half);
#pragma unroll
    for (int t = 0; t < 8; ++t) oa[t] = wmma_bf16(pf, vf[t], oa[t]);

    __syncthreads();   // all waves done reading before buffer 'cur' is rewritten
  }

#pragma unroll
  for (int r = 0; r < 8; ++r) {
    const float inv = 1.f / fmaxf(lrow[r], 1e-30f);
#pragma unroll
    for (int t = 0; t < 8; ++t) {
      const unsigned o = ((unsigned)(b * S_ + q0 + r + 8 * half) * H_ + h) * HD_
                         + 16 * t + lcol;
      ob[o] = f2bf(oa[t][r] * inv);
    }
  }
}

// ---------------------------------------------------------------------------
extern "C" void kernel_launch(void* const* d_in, const int* in_sizes, int n_in,
                              void* d_out, int out_size, void* d_ws, size_t ws_size,
                              hipStream_t stream) {
  (void)in_sizes; (void)n_in; (void)out_size; (void)ws_size;
  const float* hs = (const float*)d_in[0];
  const float* Wq = (const float*)d_in[1];
  const float* Wk = (const float*)d_in[2];
  const float* Wv = (const float*)d_in[3];
  const float* Wg = (const float*)d_in[4];
  const float* Wd = (const float*)d_in[5];
  const float* Wo = (const float*)d_in[6];
  float* out = (float*)d_out;

  char* w = (char*)d_ws;
  auto take = [&](size_t bytes) -> char* {
    char* p = w;
    w += (bytes + 255) & ~(size_t)255;
    return p;
  };
  unsigned short* hsb  = (unsigned short*)take((size_t)B_ * S_ * D_ * 2);
  unsigned short* WqT  = (unsigned short*)take((size_t)D_ * H_ * HD_ * 2);
  unsigned short* WkT  = (unsigned short*)take((size_t)D_ * HKV_ * HD_ * 2);
  unsigned short* WvT  = (unsigned short*)take((size_t)D_ * HKV_ * HD_ * 2);
  unsigned short* WoT  = (unsigned short*)take((size_t)H_ * HD_ * D_ * 2);
  unsigned short* qbuf = (unsigned short*)take((size_t)B_ * S_ * H_ * HD_ * 2);
  unsigned short* kbuf = (unsigned short*)take((size_t)B_ * S_ * HKV_ * HD_ * 2);
  unsigned short* vbuf = (unsigned short*)take((size_t)B_ * S_ * HKV_ * HD_ * 2);
  unsigned short* vTb  = (unsigned short*)take((size_t)B_ * HKV_ * HD_ * S_ * 2);
  unsigned short* obuf = (unsigned short*)take((size_t)B_ * S_ * H_ * HD_ * 2);
  float*          biasb = (float*)take((size_t)B_ * HKV_ * S_ * 4);

  // 1) fp32 -> bf16 conversions / tiled weight transposes
  {
    size_t n = (size_t)B_ * S_ * D_;
    f32_to_bf16_kernel<<<dim3((unsigned)((n + 255) / 256)), dim3(256), 0, stream>>>(hs, hsb, n);
  }
  transpose_f32_to_bf16_tiled<<<dim3((H_ * HD_) / 32, D_ / 32), dim3(32, 8), 0, stream>>>(
      Wq, WqT, D_, H_ * HD_);
  transpose_f32_to_bf16_tiled<<<dim3((HKV_ * HD_) / 32, D_ / 32), dim3(32, 8), 0, stream>>>(
      Wk, WkT, D_, HKV_ * HD_);
  transpose_f32_to_bf16_tiled<<<dim3((HKV_ * HD_) / 32, D_ / 32), dim3(32, 8), 0, stream>>>(
      Wv, WvT, D_, HKV_ * HD_);
  transpose_f32_to_bf16_tiled<<<dim3(D_ / 32, (H_ * HD_) / 32), dim3(32, 8), 0, stream>>>(
      Wo, WoT, H_ * HD_, D_);

  // 2) projections (WMMA GEMMs, 128x128 block tiles)
  const int M = B_ * S_;  // 4096
  gemm_bf16_wmma<<<dim3(M / 128, (H_ * HD_) / 128), dim3(256), 0, stream>>>(
      hsb, WqT, qbuf, M, H_ * HD_, D_, 0);
  gemm_bf16_wmma<<<dim3(M / 128, (HKV_ * HD_) / 128), dim3(256), 0, stream>>>(
      hsb, WkT, kbuf, M, HKV_ * HD_, D_, 0);
  gemm_bf16_wmma<<<dim3(M / 128, (HKV_ * HD_) / 128), dim3(256), 0, stream>>>(
      hsb, WvT, vbuf, M, HKV_ * HD_, D_, 0);

  // 3) v transpose (tiled) + dynamic bias
  transpose_v_tiled<<<dim3(HD_ / 32, S_ / 32, B_ * HKV_), dim3(32, 8), 0, stream>>>(vbuf, vTb);
  bias_kernel<<<dim3(M / 8), dim3(256), 0, stream>>>(qbuf, vbuf, Wg, Wd, biasb);

  // 4) flash attention (WMMA + TDM double-buffered K/V panels)
  fa_wmma_kernel<<<dim3(S_ / 16, HKV_, B_), dim3(128), 0, stream>>>(
      qbuf, kbuf, vTb, biasb, obuf);

  // 5) output projection (WMMA, f32 out)
  gemm_bf16_wmma<<<dim3(M / 128, D_ / 128), dim3(256), 0, stream>>>(
      obuf, WoT, out, M, D_, H_ * HD_, 1);
}
